// MambaMoEBlock_16870631538809
// MI455X (gfx1250) — compile-verified, matrix operand order checked by reading, not run on hardware
//
#include <hip/hip_runtime.h>
#include <hip/hip_bf16.h>

// ---------------- problem constants (match reference) ----------------
#define Bc      2
#define Lc      2048
#define DIMc    512
#define DSTATE  16
#define DCONV   4
#define DIc     1024          // D_INNER
#define TDIc    2048          // 2*D_INNER
#define DTR     32            // DT_RANK
#define Ec      8
#define HIDc    2048
#define Tc      (Bc*Lc)       // 4096 tokens
#define CHUNK   128           // HID chunk for MoE FFN
#define KMAX    1024          // max GEMM K (for LDS A-tile staging)

typedef _Float16 h16;
typedef __attribute__((ext_vector_type(16))) _Float16 v16h;
typedef __attribute__((ext_vector_type(8)))  float    v8f;

// ---------------- CDNA5 async global->LDS path (guarded) ----------------
#if defined(__HIP_DEVICE_COMPILE__) && \
    __has_builtin(__builtin_amdgcn_global_load_async_to_lds_b128) && \
    __has_builtin(__builtin_amdgcn_s_wait_asynccnt)
#define HAVE_ASYNC_LDS 1
#endif

#if HAVE_ASYNC_LDS
typedef __attribute__((__vector_size__(16))) int v4i_;
typedef __attribute__((address_space(1))) v4i_* gv4p_t;   // global v4i*
typedef __attribute__((address_space(3))) v4i_* lv4p_t;   // LDS v4i*
#endif

// copy one 16B chunk global -> LDS (async if available)
__device__ __forceinline__ void cp16_g2l(void* lds_dst, const void* gsrc) {
#if HAVE_ASYNC_LDS
    __builtin_amdgcn_global_load_async_to_lds_b128((gv4p_t)gsrc, (lv4p_t)lds_dst, 0, 0);
#else
    *(int4*)lds_dst = *(const int4*)gsrc;
#endif
}
__device__ __forceinline__ void cp16_wait() {
#if HAVE_ASYNC_LDS
    __builtin_amdgcn_s_wait_asynccnt((unsigned short)0);
#endif
}

union Frag { v16h v; int4 q[2]; };

// Load a 16-half WMMA fragment: per-lane two contiguous 16B segments at
// p[0..7] and p[16..23].  Matches ISA 16-bit A/B layout (lane=row, hi half
// of wave offsets K by 8, element i>=8 offsets K by 16).  p must be 16B aligned.
__device__ __forceinline__ v16h ldfrag(const h16* p) {
    Frag f;
    f.q[0] = *(const int4*)(p);
    f.q[1] = *(const int4*)(p + 16);
    return f.v;
}

__device__ __forceinline__ float sigmoidf_(float x) { return 1.0f / (1.0f + __expf(-x)); }
__device__ __forceinline__ float geluf_(float x) {
    // tanh-approx GELU (jax.nn.gelu default)
    float c = 0.7978845608028654f * (x + 0.044715f * x * x * x);
    return 0.5f * x * (1.0f + tanhf(c));
}

// ---------------- small utility kernels ----------------

__global__ void cvt_f32_h16(const float* __restrict__ src, h16* __restrict__ dst, int n) {
    int i = blockIdx.x * blockDim.x + threadIdx.x;
    if (i < n) dst[i] = (h16)src[i];
}

// dst[e][c][r] = src[e][r][c]   (transpose last two dims + convert to f16)
__global__ void tcvt_f32_h16(const float* __restrict__ src, h16* __restrict__ dst, int R, int C) {
    int e = blockIdx.y;
    int i = blockIdx.x * blockDim.x + threadIdx.x;       // i over R*C
    if (i >= R * C) return;
    int r = i / C, c = i % C;
    dst[(size_t)e * R * C + (size_t)c * R + r] = (h16)src[(size_t)e * R * C + (size_t)r * C + c];
}

// SimpleRMSNorm per token row of DIMc, output f16
__global__ __launch_bounds__(256) void rms_h16(const float* __restrict__ in, h16* __restrict__ out) {
    int t = blockIdx.x;
    const float* row = in + (size_t)t * DIMc;
    float ss = 0.0f;
    for (int i = threadIdx.x; i < DIMc; i += 256) { float v = row[i]; ss += v * v; }
    __shared__ float red[8];
    __shared__ float scl;
    for (int o = 16; o > 0; o >>= 1) ss += __shfl_down(ss, o);
    if ((threadIdx.x & 31) == 0) red[threadIdx.x >> 5] = ss;
    __syncthreads();
    if (threadIdx.x == 0) {
        float s = 0.0f;
        #pragma unroll
        for (int i = 0; i < 8; i++) s += red[i];
        scl = rsqrtf(s / (float)DIMc + 1e-6f);
    }
    __syncthreads();
    float sc = scl;
    for (int i = threadIdx.x; i < DIMc; i += 256) out[(size_t)t * DIMc + i] = (h16)(row[i] * sc);
}

// ---------------- generic WMMA GEMM: C[T,N] = A[T,K] * W[N,K]^T (+resid) ----------------
// block = 256 threads = 8 waves; wave computes one 16x16 C tile.
// A tile (16 consecutive rows = one contiguous 32KB-max block) is staged once
// per block into LDS via the async global->LDS path and shared by all 8 waves;
// the B (weight) operand streams from L2 with prefetch.
__global__ __launch_bounds__(256) void gemm_h16(const h16* __restrict__ A,
                                                const h16* __restrict__ W,
                                                float* __restrict__ C,
                                                const float* __restrict__ resid,
                                                int N, int K) {
    __shared__ __align__(16) h16 Asm[16 * KMAX];   // 32 KB
    int wave = threadIdx.x >> 5, lane = threadIdx.x & 31, tid = threadIdx.x;
    int mt = blockIdx.y;
    int nt = blockIdx.x * 8 + wave;

    // ---- stage A tile: 16 rows x K halfs, contiguous ----
    {
        int nchunk = (16 * K) >> 3;                 // 16B chunks
        const char* Ab = (const char*)(A + (size_t)mt * 16 * K);
        char* Lb = (char*)Asm;
        for (int c = tid; c < nchunk; c += 256)
            cp16_g2l(Lb + (size_t)c * 16, Ab + (size_t)c * 16);
        cp16_wait();
    }
    __syncthreads();
    if (nt * 16 >= N) return;

    int hi = (lane >> 4) & 1;
    int nl = lane & 15;
    const h16* Ap = Asm + (size_t)nl * K + 8 * hi;                  // LDS
    const h16* Wp = W + (size_t)(nt * 16 + nl) * K + 8 * hi;        // global
    v8f acc = {};
    for (int k0 = 0; k0 < K; k0 += 32) {
        __builtin_prefetch(Wp + k0 + 256, 0, 1);   // global_prefetch_b8
        v16h a = ldfrag(Ap + k0);
        v16h b = ldfrag(Wp + k0);
        acc = __builtin_amdgcn_wmma_f32_16x16x32_f16(false, a, false, b, (short)0, acc, false, false);
    }
    int cn = nt * 16 + nl;
    #pragma unroll
    for (int r = 0; r < 8; r++) {
        int m = mt * 16 + r + 8 * hi;
        float v = acc[r];
        if (resid) v += resid[(size_t)m * N + cn];
        C[(size_t)m * N + cn] = v;
    }
}

// ---------------- Mamba pieces ----------------

// causal depthwise conv (width 4) + bias + SiLU; reads xm from first half of xr
__global__ void conv_silu_k(const float* __restrict__ xr, const float* __restrict__ conv_w,
                            const float* __restrict__ conv_b,
                            float* __restrict__ xm_f, h16* __restrict__ xm_h) {
    int idx = blockIdx.x * blockDim.x + threadIdx.x;    // over Tc*DIc
    if (idx >= Tc * DIc) return;
    int t = idx / DIc, d = idx % DIc;
    int b = t / Lc, l = t % Lc;
    float acc = conv_b[d];
    #pragma unroll
    for (int j = 0; j < DCONV; j++) {
        int ls = l - (DCONV - 1) + j;
        if (ls >= 0) acc += conv_w[d * DCONV + j] * xr[(size_t)(b * Lc + ls) * TDIc + d];
    }
    float s = acc * sigmoidf_(acc);
    xm_f[idx] = s;
    xm_h[idx] = (h16)s;
}

// delta = softplus(dt @ dt_proj_w^T + dt_proj_b)
__global__ void delta_k(const float* __restrict__ xdbl, const float* __restrict__ dt_w,
                        const float* __restrict__ dt_b, float* __restrict__ delta) {
    int idx = blockIdx.x * blockDim.x + threadIdx.x;    // over Tc*DIc
    if (idx >= Tc * DIc) return;
    int t = idx / DIc, d = idx % DIc;
    float a = dt_b[d];
    const float* xd = xdbl + (size_t)t * 64;
    #pragma unroll
    for (int r = 0; r < DTR; r++) a += xd[r] * dt_w[d * DTR + r];
    delta[idx] = (a > 20.0f) ? a : log1pf(__expf(a));
}

// selective scan: one thread per (b, d); 16-state recurrence in registers.
// Fuses +u*D and the SiLU(res) gate, writes f16 for out_proj GEMM.
__global__ void scan_k(const float* __restrict__ delta, const float* __restrict__ xm_f,
                       const float* __restrict__ xdbl, const float* __restrict__ A_log,
                       const float* __restrict__ Dp, const float* __restrict__ xr,
                       h16* __restrict__ y_h) {
    int idx = blockIdx.x * blockDim.x + threadIdx.x;    // 0 .. Bc*DIc-1
    if (idx >= Bc * DIc) return;
    int b = idx / DIc, d = idx % DIc;
    float Aco[DSTATE];
    #pragma unroll
    for (int n = 0; n < DSTATE; n++) Aco[n] = -__expf(A_log[d * DSTATE + n]);
    float Dv = Dp[d];
    float s[DSTATE];
    #pragma unroll
    for (int n = 0; n < DSTATE; n++) s[n] = 0.0f;
    for (int l = 0; l < Lc; l++) {
        int t = b * Lc + l;
        float dlt = delta[(size_t)t * DIc + d];
        float u   = xm_f[(size_t)t * DIc + d];
        const float* xd = xdbl + (size_t)t * 64;
        float y = 0.0f;
        #pragma unroll
        for (int n = 0; n < DSTATE; n++) {
            float dA = __expf(dlt * Aco[n]);
            s[n] = dA * s[n] + dlt * xd[32 + n] * u;
            y += s[n] * xd[48 + n];
        }
        float rv = xr[(size_t)t * TDIc + DIc + d];
        float gate = rv * sigmoidf_(rv);
        y_h[(size_t)t * DIc + d] = (h16)((y + u * Dv) * gate);
    }
}

// ---------------- MoE routing (deterministic, no atomics) ----------------

__global__ void gate_topk_k(const h16* __restrict__ xn2, const float* __restrict__ gate_w,
                            int* __restrict__ topi, float* __restrict__ topg) {
    int warp = threadIdx.x >> 5, lane = threadIdx.x & 31;
    int t = blockIdx.x * 8 + warp;
    if (t >= Tc) return;
    float lg[Ec];
    #pragma unroll
    for (int e = 0; e < Ec; e++) lg[e] = 0.0f;
    for (int d = lane; d < DIMc; d += 32) {
        float xv = (float)xn2[(size_t)t * DIMc + d];
        #pragma unroll
        for (int e = 0; e < Ec; e++) lg[e] += xv * gate_w[e * DIMc + d];
    }
    #pragma unroll
    for (int e = 0; e < Ec; e++)
        for (int o = 16; o > 0; o >>= 1) lg[e] += __shfl_down(lg[e], o);
    if (lane == 0) {
        float mx = lg[0];
        #pragma unroll
        for (int e = 1; e < Ec; e++) mx = fmaxf(mx, lg[e]);
        float p[Ec], se = 0.0f;
        #pragma unroll
        for (int e = 0; e < Ec; e++) { p[e] = __expf(lg[e] - mx); se += p[e]; }
        #pragma unroll
        for (int e = 0; e < Ec; e++) p[e] /= se;
        int i0 = 0;
        for (int e = 1; e < Ec; e++) if (p[e] > p[i0]) i0 = e;
        int i1 = (i0 == 0) ? 1 : 0;
        for (int e = 0; e < Ec; e++) if (e != i0 && p[e] > p[i1]) i1 = e;
        float s2 = p[i0] + p[i1] + 1e-9f;
        topi[t * 2 + 0] = i0;  topg[t * 2 + 0] = p[i0] / s2;
        topi[t * 2 + 1] = i1;  topg[t * 2 + 1] = p[i1] / s2;
    }
}

// one thread per expert scans tokens in order -> deterministic compact lists
__global__ void build_lists_k(const int* __restrict__ topi, const float* __restrict__ topg,
                              int* __restrict__ cnt, int* __restrict__ list,
                              int* __restrict__ slot, float* __restrict__ gv) {
    int e = threadIdx.x;
    if (e >= Ec) return;
    int c = 0;
    for (int t = 0; t < Tc; t++) {
        #pragma unroll
        for (int k = 0; k < 2; k++) {
            if (topi[t * 2 + k] == e) {
                list[e * Tc + c] = t;
                slot[e * Tc + c] = k;
                gv[e * Tc + c]   = topg[t * 2 + k];
                c++;
            }
        }
    }
    cnt[e] = c;
}

// ---------------- MoE expert FFN (gathered, WMMA, LDS-chunked) ----------------
// block = (expert e, token-tile); 8 waves; out contribution scattered to
// obuf[(token*2 + slot)*DIM + n]  (each (token,slot) written exactly once).
__global__ __launch_bounds__(256) void moe_ffn_k(const h16* __restrict__ xn2,
                                                 const h16* __restrict__ w1t,
                                                 const h16* __restrict__ w2t,
                                                 const float* __restrict__ b1,
                                                 const float* __restrict__ b2,
                                                 const int* __restrict__ cnt,
                                                 const int* __restrict__ list,
                                                 const int* __restrict__ slot,
                                                 const float* __restrict__ gv,
                                                 float* __restrict__ obuf) {
    int e = blockIdx.y, tile = blockIdx.x;
    int n_e = cnt[e];
    if (tile * 16 >= n_e) return;

    __shared__ int   toks[16];
    __shared__ int   slots[16];
    __shared__ float gvs[16];
    __shared__ __align__(16) h16 Atile[16][DIMc];   // 16 KB
    __shared__ __align__(16) h16 Htile[16][CHUNK];  //  4 KB

    int tid = threadIdx.x, wave = tid >> 5, lane = tid & 31;
    int hi = (lane >> 4) & 1, nl = lane & 15;

    if (tid < 16) {
        int pos = tile * 16 + tid;
        bool valid = pos < n_e;
        toks[tid]  = valid ? list[e * Tc + pos] : 0;
        slots[tid] = valid ? slot[e * Tc + pos] : -1;
        gvs[tid]   = valid ? gv[e * Tc + pos]   : 0.0f;
    }
    __syncthreads();

    // gather 16 token rows of xn2 into LDS (async global->LDS, 4x16B per thread)
    {
        int r  = tid >> 4;
        int c0 = (tid & 15) * 32;
        const h16* src = xn2 + (size_t)toks[r] * DIMc + c0;
        #pragma unroll
        for (int j = 0; j < 4; j++)
            cp16_g2l(&Atile[r][c0 + j * 8], src + j * 8);
        cp16_wait();
    }
    __syncthreads();

    v8f acc[4];
    #pragma unroll
    for (int j = 0; j < 4; j++) { v8f z = {}; acc[j] = z; }

    for (int c = 0; c < HIDc; c += CHUNK) {
        // GEMM1: h[16 x CHUNK] = Atile[16x512] @ w1[e][512 x CHUNK]; wave -> 16 cols
        v8f hacc = {};
        const h16* Bp = w1t + (size_t)e * HIDc * DIMc + (size_t)(c + wave * 16 + nl) * DIMc + 8 * hi;
        const h16* Ap = &Atile[nl][8 * hi];
        #pragma unroll
        for (int k0 = 0; k0 < DIMc; k0 += 32) {
            v16h a = ldfrag(Ap + k0);
            v16h b = ldfrag(Bp + k0);
            hacc = __builtin_amdgcn_wmma_f32_16x16x32_f16(false, a, false, b, (short)0, hacc, false, false);
        }
        // bias + GELU -> LDS (f16)
        {
            int hidx = c + wave * 16 + nl;
            float bb = b1[e * HIDc + hidx];
            #pragma unroll
            for (int r = 0; r < 8; r++) {
                int m = r + 8 * hi;
                Htile[m][wave * 16 + nl] = (h16)geluf_(hacc[r] + bb);
            }
        }
        __syncthreads();
        // GEMM2: acc[16 x 512] += Htile[16 x CHUNK] @ w2[e][CHUNK x 512]; wave -> 4 n-tiles
        #pragma unroll
        for (int j = 0; j < 4; j++) {
            int nb = (wave * 4 + j) * 16;
            const h16* Wp = w2t + (size_t)e * DIMc * HIDc + (size_t)(nb + nl) * HIDc + c + 8 * hi;
            const h16* Hp = &Htile[nl][8 * hi];
            #pragma unroll
            for (int k0 = 0; k0 < CHUNK; k0 += 32) {
                v16h a = ldfrag(Hp + k0);
                v16h b = ldfrag(Wp + k0);
                acc[j] = __builtin_amdgcn_wmma_f32_16x16x32_f16(false, a, false, b, (short)0, acc[j], false, false);
            }
        }
        __syncthreads();
    }

    // epilogue: gated scatter to per-(token,slot) buffer (deterministic)
    #pragma unroll
    for (int j = 0; j < 4; j++) {
        int nb = (wave * 4 + j) * 16 + nl;
        float bb = b2[e * DIMc + nb];
        #pragma unroll
        for (int r = 0; r < 8; r++) {
            int m = r + 8 * hi;
            if (slots[m] >= 0) {
                float v = (acc[j][r] + bb) * gvs[m];
                obuf[((size_t)toks[m] * 2 + slots[m]) * DIMc + nb] = v;
            }
        }
    }
}

__global__ void final_add_k(const float* __restrict__ x2, const float* __restrict__ obuf,
                            float* __restrict__ out) {
    int i = blockIdx.x * blockDim.x + threadIdx.x;      // over Tc*DIMc
    if (i >= Tc * DIMc) return;
    int t = i / DIMc, n = i % DIMc;
    out[i] = x2[i] + obuf[((size_t)t * 2 + 0) * DIMc + n] + obuf[((size_t)t * 2 + 1) * DIMc + n];
}

// ---------------- host orchestration ----------------

extern "C" void kernel_launch(void* const* d_in, const int* in_sizes, int n_in,
                              void* d_out, int out_size, void* d_ws, size_t ws_size,
                              hipStream_t stream) {
    (void)in_sizes; (void)n_in; (void)out_size; (void)ws_size;

    const float* x        = (const float*)d_in[0];
    const float* in_pw    = (const float*)d_in[1];
    const float* conv_w   = (const float*)d_in[2];
    const float* conv_b   = (const float*)d_in[3];
    const float* x_pw     = (const float*)d_in[4];
    const float* dt_pw    = (const float*)d_in[5];
    const float* dt_pb    = (const float*)d_in[6];
    const float* A_log    = (const float*)d_in[7];
    const float* Dp       = (const float*)d_in[8];
    const float* out_pw   = (const float*)d_in[9];
    const float* gate_w   = (const float*)d_in[10];
    const float* w1       = (const float*)d_in[11];
    const float* b1       = (const float*)d_in[12];
    const float* w2       = (const float*)d_in[13];
    const float* b2       = (const float*)d_in[14];
    float* out = (float*)d_out;

    char* w = (char*)d_ws;
    auto alloc = [&](size_t bytes) -> void* {
        void* p = (void*)w;
        w += (bytes + 255) & ~(size_t)255;
        return p;
    };

    h16*   w_inproj = (h16*)  alloc((size_t)TDIc * DIMc * 2);
    h16*   w_xproj  = (h16*)  alloc((size_t)(DTR + 2 * DSTATE) * DIc * 2);
    h16*   w_outprj = (h16*)  alloc((size_t)DIMc * DIc * 2);
    h16*   w1t      = (h16*)  alloc((size_t)Ec * DIMc * HIDc * 2);
    h16*   w2t      = (h16*)  alloc((size_t)Ec * HIDc * DIMc * 2);
    h16*   xn_h     = (h16*)  alloc((size_t)Tc * DIMc * 2);
    float* xr       = (float*)alloc((size_t)Tc * TDIc * 4);
    float* xm_f     = (float*)alloc((size_t)Tc * DIc * 4);
    h16*   xm_h     = (h16*)  alloc((size_t)Tc * DIc * 2);
    float* xdbl     = (float*)alloc((size_t)Tc * 64 * 4);
    float* deltab   = (float*)alloc((size_t)Tc * DIc * 4);
    h16*   y_h      = (h16*)  alloc((size_t)Tc * DIc * 2);
    float* x2       = (float*)alloc((size_t)Tc * DIMc * 4);
    h16*   xn2_h    = (h16*)  alloc((size_t)Tc * DIMc * 2);
    int*   topi     = (int*)  alloc((size_t)Tc * 2 * 4);
    float* topg     = (float*)alloc((size_t)Tc * 2 * 4);
    int*   cnt      = (int*)  alloc(Ec * 4);
    int*   list     = (int*)  alloc((size_t)Ec * Tc * 4);
    int*   slotb    = (int*)  alloc((size_t)Ec * Tc * 4);
    float* gvb      = (float*)alloc((size_t)Ec * Tc * 4);
    float* obuf     = (float*)alloc((size_t)Tc * 2 * DIMc * 4);

    // 1. weight conversions / transposes (f32 -> f16 WMMA operands)
    cvt_f32_h16<<<(TDIc * DIMc + 255) / 256, 256, 0, stream>>>(in_pw, w_inproj, TDIc * DIMc);
    cvt_f32_h16<<<((DTR + 2 * DSTATE) * DIc + 255) / 256, 256, 0, stream>>>(x_pw, w_xproj, (DTR + 2 * DSTATE) * DIc);
    cvt_f32_h16<<<(DIMc * DIc + 255) / 256, 256, 0, stream>>>(out_pw, w_outprj, DIMc * DIc);
    tcvt_f32_h16<<<dim3((DIMc * HIDc + 255) / 256, Ec), 256, 0, stream>>>(w1, w1t, DIMc, HIDc);
    tcvt_f32_h16<<<dim3((HIDc * DIMc + 255) / 256, Ec), 256, 0, stream>>>(w2, w2t, HIDc, DIMc);

    // 2. Mamba branch
    rms_h16<<<Tc, 256, 0, stream>>>(x, xn_h);
    gemm_h16<<<dim3(TDIc / 128, Tc / 16), 256, 0, stream>>>(xn_h, w_inproj, xr, nullptr, TDIc, DIMc);
    conv_silu_k<<<(Tc * DIc + 255) / 256, 256, 0, stream>>>(xr, conv_w, conv_b, xm_f, xm_h);
    gemm_h16<<<dim3(1, Tc / 16), 256, 0, stream>>>(xm_h, w_xproj, xdbl, nullptr, DTR + 2 * DSTATE, DIc);
    delta_k<<<(Tc * DIc + 255) / 256, 256, 0, stream>>>(xdbl, dt_pw, dt_pb, deltab);
    scan_k<<<(Bc * DIc + 255) / 256, 256, 0, stream>>>(deltab, xm_f, xdbl, A_log, Dp, xr, y_h);
    gemm_h16<<<dim3(DIMc / 128, Tc / 16), 256, 0, stream>>>(y_h, w_outprj, x2, x, DIMc, DIc);

    // 3. MoE branch (routed top-2, deterministic)
    rms_h16<<<Tc, 256, 0, stream>>>(x2, xn2_h);
    gate_topk_k<<<Tc / 8, 256, 0, stream>>>(xn2_h, gate_w, topi, topg);
    build_lists_k<<<1, 32, 0, stream>>>(topi, topg, cnt, list, slotb, gvb);
    moe_ffn_k<<<dim3(Tc / 16, Ec), 256, 0, stream>>>(xn2_h, w1t, w2t, b1, b2, cnt, list, slotb, gvb, obuf);
    final_add_k<<<(Tc * DIMc + 255) / 256, 256, 0, stream>>>(x2, obuf, out);
}
